// MMG_2190433321478
// MI455X (gfx1250) — matrix operand order repaired and checked by reading, not compiled, not run on hardware
//
#include <hip/hip_runtime.h>
#include <hip/hip_bf16.h>
#include <math.h>

typedef __attribute__((ext_vector_type(16))) _Float16 v16h;
typedef __attribute__((ext_vector_type(8)))  float    v8f;

// ---------------------------------------------------------------------------
// WMMA helpers (CDNA5 gfx1250, wave32)
// ---------------------------------------------------------------------------
__device__ __forceinline__ v8f wmma_f16(v16h a, v16h b, v8f c) {
    // D = A(16x32 f16) * B(32x16 f16) + C(16x16 f32)
    return __builtin_amdgcn_wmma_f32_16x16x32_f16(
        /*neg_a=*/false, a, /*neg_b=*/false, b,
        /*c_mod=*/(short)0, c, /*reuse_a=*/false, /*reuse_b=*/false);
}

// A-fragment (16x32 f16) from row-major f16 LDS buffer [rows][stride].
// ISA layout: lane L -> row = L&15, half h = L>>4.
//   VGPR q<4 : K = k0 + 8h + 2q,  VGPR q>=4 : K = k0 + 16 + 8h + 2(q-4) = k0+8h+8+2q
__device__ __forceinline__ v16h load_a_frag(const _Float16* base, int stride,
                                            int row0, int k0) {
    const int lane = threadIdx.x & 31;
    const int r = lane & 15, hh = lane >> 4;
    const _Float16* row = base + (size_t)(row0 + r) * stride + k0 + 8 * hh;
    union { v16h v; unsigned u[8]; } a;
#pragma unroll
    for (int q = 0; q < 8; ++q) {
        const int k = (q < 4) ? (2 * q) : (8 + 2 * q);
        a.u[q] = *(const unsigned*)(row + k);   // 2 consecutive f16
    }
    return a.v;
}

// B-fragment from pre-packed weights: [kt][nt][lane][16 f16] contiguous.
__device__ __forceinline__ v16h load_b_frag(const _Float16* P, int ntn,
                                            int kt, int nt) {
    const int lane = threadIdx.x & 31;
    const v16h* p = (const v16h*)(P + ((size_t)(kt * ntn + nt) * 32 + lane) * 16);
    return *p;
}

// ---------------------------------------------------------------------------
// Weight packing: W[K,N] f32 row-major -> WMMA B-fragment tiles in f16.
// ISA B layout (32x16 f16): lane L -> col = L&15, half h = L>>4,
//   element i (=2v+e) holds K = 8h + i (i<8), 16 + 8h + (i-8) (i>=8).
// ---------------------------------------------------------------------------
__global__ void pack_weight_kernel(const float* __restrict__ W,
                                   _Float16* __restrict__ P, int K, int N) {
    const int idx = blockIdx.x * blockDim.x + threadIdx.x;
    if (idx >= K * N) return;
    const int i    = idx & 15;
    const int lane = (idx >> 4) & 31;
    const int tile = idx >> 9;
    const int ntn  = N >> 4;
    const int kt = tile / ntn, nt = tile % ntn;
    const int hh = lane >> 4, c = lane & 15;
    const int k = kt * 32 + (i < 8 ? hh * 8 + i : 16 + hh * 8 + (i - 8));
    const int n = nt * 16 + c;
    P[idx] = (_Float16)W[(size_t)k * N + n];
}

__global__ void zero_f32_kernel(float* __restrict__ p, int n) {
    int i = blockIdx.x * blockDim.x + threadIdx.x;
    if (i < n) p[i] = 0.0f;
}

// ---------------------------------------------------------------------------
// EdgeConv layer 1: F=128.  m = [x_dst | x_src - x_dst] (32 edges x 256)
//   GEMM1: [32,256] @ W1a[256,512] -> relu -> LDS
//   GEMM2: [32,512] @ W2a[512,256] -> relu -> atomicMax into agg1[N,256]
// ---------------------------------------------------------------------------
#define SM1_STRIDE 264   // 256 + 8 pad (f16)  -> 132 dwords, bank stride 4
#define SH1_STRIDE 520   // 512 + 8 pad (f16)  -> 260 dwords, bank stride 4

__global__ __launch_bounds__(256) void edgeconv1_kernel(
    const float* __restrict__ x, const int* __restrict__ ei, int E,
    const _Float16* __restrict__ pW1, const float* __restrict__ b1,
    const _Float16* __restrict__ pW2, const float* __restrict__ b2,
    float* __restrict__ agg) {
    __shared__ _Float16 sm[32 * SM1_STRIDE];
    __shared__ _Float16 sh[32 * SH1_STRIDE];
    __shared__ int sdst[32];

    const int t = threadIdx.x;
    const int e0 = blockIdx.x * 32;

    // ---- gather: 8 threads/edge, 16 features each (of 128) ----
    {
        const int e = t >> 3, q = t & 7;
        const int ge = e0 + e;
        int d = -1;
        if (ge < E) {
            const int s = ei[ge];
            d = ei[E + ge];
            const float4* xi = (const float4*)(x + (size_t)d * 128) + q * 4;
            const float4* xj = (const float4*)(x + (size_t)s * 128) + q * 4;
            _Float16* mr = sm + e * SM1_STRIDE;
#pragma unroll
            for (int i = 0; i < 4; ++i) {
                const float4 ai = xi[i], aj = xj[i];
                const int f = q * 16 + i * 4;
                mr[f + 0] = (_Float16)ai.x; mr[f + 1] = (_Float16)ai.y;
                mr[f + 2] = (_Float16)ai.z; mr[f + 3] = (_Float16)ai.w;
                mr[128 + f + 0] = (_Float16)(aj.x - ai.x);
                mr[128 + f + 1] = (_Float16)(aj.y - ai.y);
                mr[128 + f + 2] = (_Float16)(aj.z - ai.z);
                mr[128 + f + 3] = (_Float16)(aj.w - ai.w);
            }
        }
        if (q == 0) sdst[e] = d;
    }
    __syncthreads();

    const int wave = t >> 5, lane = t & 31;
    const int col = lane & 15, hh = lane >> 4;

    // ---- GEMM1: 32 N-strips, 4 per wave; 2 M-tiles share each B fragment ----
    for (int s = 0; s < 4; ++s) {
        const int nt = wave * 4 + s;           // N tile of 512
        v8f acc0 = {}, acc1 = {};
        for (int kt = 0; kt < 8; ++kt) {       // K = 256
            const v16h b  = load_b_frag(pW1, 32, kt, nt);
            const v16h a0 = load_a_frag(sm, SM1_STRIDE, 0, kt * 32);
            const v16h a1 = load_a_frag(sm, SM1_STRIDE, 16, kt * 32);
            acc0 = wmma_f16(a0, b, acc0);
            acc1 = wmma_f16(a1, b, acc1);
        }
        const float bias = b1[nt * 16 + col];
#pragma unroll
        for (int v = 0; v < 8; ++v) {
            const int rr = v + 8 * hh;
            sh[(size_t)rr * SH1_STRIDE + nt * 16 + col] =
                (_Float16)fmaxf(acc0[v] + bias, 0.0f);
            sh[(size_t)(16 + rr) * SH1_STRIDE + nt * 16 + col] =
                (_Float16)fmaxf(acc1[v] + bias, 0.0f);
        }
    }
    __syncthreads();

    // ---- GEMM2: 16 N-strips, 2 per wave ----
    for (int s = 0; s < 2; ++s) {
        const int nt = wave * 2 + s;           // N tile of 256
        v8f acc0 = {}, acc1 = {};
        for (int kt = 0; kt < 16; ++kt) {      // K = 512
            const v16h b  = load_b_frag(pW2, 16, kt, nt);
            const v16h a0 = load_a_frag(sh, SH1_STRIDE, 0, kt * 32);
            const v16h a1 = load_a_frag(sh, SH1_STRIDE, 16, kt * 32);
            acc0 = wmma_f16(a0, b, acc0);
            acc1 = wmma_f16(a1, b, acc1);
        }
        const float bias = b2[nt * 16 + col];
#pragma unroll
        for (int v = 0; v < 8; ++v) {
            const int r0 = v + 8 * hh, r1 = 16 + r0;
            const float v0 = fmaxf(acc0[v] + bias, 0.0f);
            const float v1 = fmaxf(acc1[v] + bias, 0.0f);
            const int n0 = sdst[r0], n1 = sdst[r1];
            if (n0 >= 0)
                atomicMax((unsigned int*)&agg[(size_t)n0 * 256 + nt * 16 + col],
                          __float_as_uint(v0));
            if (n1 >= 0)
                atomicMax((unsigned int*)&agg[(size_t)n1 * 256 + nt * 16 + col],
                          __float_as_uint(v1));
        }
    }
}

// ---------------------------------------------------------------------------
// EdgeConv layer 2: F=256.  m = [h_dst | h_src - h_dst] (32 edges x 512)
//   GEMM1: [32,512] @ W1b[512,256] -> relu -> LDS
//   GEMM2: [32,256] @ W2b[256,128] -> relu -> atomicMax into agg2[N,128]
// ---------------------------------------------------------------------------
__global__ __launch_bounds__(256) void edgeconv2_kernel(
    const float* __restrict__ hin, const int* __restrict__ ei, int E,
    const _Float16* __restrict__ pW1, const float* __restrict__ b1,
    const _Float16* __restrict__ pW2, const float* __restrict__ b2,
    float* __restrict__ agg) {
    __shared__ _Float16 sm[32 * SH1_STRIDE];   // m: [32,512] stride 520
    __shared__ _Float16 sh[32 * SM1_STRIDE];   // h: [32,256] stride 264
    __shared__ int sdst[32];

    const int t = threadIdx.x;
    const int e0 = blockIdx.x * 32;

    // ---- gather: 8 threads/edge, 32 features each (of 256) ----
    {
        const int e = t >> 3, q = t & 7;
        const int ge = e0 + e;
        int d = -1;
        if (ge < E) {
            const int s = ei[ge];
            d = ei[E + ge];
            const float4* hi = (const float4*)(hin + (size_t)d * 256) + q * 8;
            const float4* hj = (const float4*)(hin + (size_t)s * 256) + q * 8;
            _Float16* mr = sm + e * SH1_STRIDE;
#pragma unroll
            for (int i = 0; i < 8; ++i) {
                const float4 ai = hi[i], aj = hj[i];
                const int f = q * 32 + i * 4;
                mr[f + 0] = (_Float16)ai.x; mr[f + 1] = (_Float16)ai.y;
                mr[f + 2] = (_Float16)ai.z; mr[f + 3] = (_Float16)ai.w;
                mr[256 + f + 0] = (_Float16)(aj.x - ai.x);
                mr[256 + f + 1] = (_Float16)(aj.y - ai.y);
                mr[256 + f + 2] = (_Float16)(aj.z - ai.z);
                mr[256 + f + 3] = (_Float16)(aj.w - ai.w);
            }
        }
        if (q == 0) sdst[e] = d;
    }
    __syncthreads();

    const int wave = t >> 5, lane = t & 31;
    const int col = lane & 15, hh = lane >> 4;

    // ---- GEMM1: [32,512] x [512,256]; 16 strips, 2 per wave ----
    for (int s = 0; s < 2; ++s) {
        const int nt = wave * 2 + s;
        v8f acc0 = {}, acc1 = {};
        for (int kt = 0; kt < 16; ++kt) {
            const v16h b  = load_b_frag(pW1, 16, kt, nt);
            const v16h a0 = load_a_frag(sm, SH1_STRIDE, 0, kt * 32);
            const v16h a1 = load_a_frag(sm, SH1_STRIDE, 16, kt * 32);
            acc0 = wmma_f16(a0, b, acc0);
            acc1 = wmma_f16(a1, b, acc1);
        }
        const float bias = b1[nt * 16 + col];
#pragma unroll
        for (int v = 0; v < 8; ++v) {
            const int rr = v + 8 * hh;
            sh[(size_t)rr * SM1_STRIDE + nt * 16 + col] =
                (_Float16)fmaxf(acc0[v] + bias, 0.0f);
            sh[(size_t)(16 + rr) * SM1_STRIDE + nt * 16 + col] =
                (_Float16)fmaxf(acc1[v] + bias, 0.0f);
        }
    }
    __syncthreads();

    // ---- GEMM2: [32,256] x [256,128]; 8 strips, 1 per wave ----
    {
        const int nt = wave;
        v8f acc0 = {}, acc1 = {};
        for (int kt = 0; kt < 8; ++kt) {
            const v16h b  = load_b_frag(pW2, 8, kt, nt);
            const v16h a0 = load_a_frag(sh, SM1_STRIDE, 0, kt * 32);
            const v16h a1 = load_a_frag(sh, SM1_STRIDE, 16, kt * 32);
            acc0 = wmma_f16(a0, b, acc0);
            acc1 = wmma_f16(a1, b, acc1);
        }
        const float bias = b2[nt * 16 + col];
#pragma unroll
        for (int v = 0; v < 8; ++v) {
            const int r0 = v + 8 * hh, r1 = 16 + r0;
            const float v0 = fmaxf(acc0[v] + bias, 0.0f);
            const float v1 = fmaxf(acc1[v] + bias, 0.0f);
            const int n0 = sdst[r0], n1 = sdst[r1];
            if (n0 >= 0)
                atomicMax((unsigned int*)&agg[(size_t)n0 * 128 + nt * 16 + col],
                          __float_as_uint(v0));
            if (n1 >= 0)
                atomicMax((unsigned int*)&agg[(size_t)n1 * 128 + nt * 16 + col],
                          __float_as_uint(v1));
        }
    }
}

// ---------------------------------------------------------------------------
// Head: y = relu(agg2 @ W3 + b3) [N,64]; out = sigmoid(y @ W4 + b4) [N,1]
// One wave per 16-node tile (WMMA), then shfl reduction for the 64->1 dot.
// ---------------------------------------------------------------------------
#define HA_STRIDE 136    // 128 + 8 pad (f16) -> 68 dwords, bank stride 4

__global__ __launch_bounds__(256) void head_kernel(
    const float* __restrict__ h, const _Float16* __restrict__ pW3,
    const float* __restrict__ b3, const float* __restrict__ W4,
    const float* __restrict__ b4, float* __restrict__ out, int Nn) {
    __shared__ _Float16 sA[8 * 16 * HA_STRIDE];

    const int t = threadIdx.x, wave = t >> 5, lane = t & 31;
    const int tile = blockIdx.x * 8 + wave;
    const int base = tile * 16;
    const bool active = (base + 15) < Nn || base < Nn;  // tiles align (N%16==0)
    _Float16* my = sA + wave * 16 * HA_STRIDE;

    if (active) {
        // stage 16x128 f32 -> f16 LDS (16 float4 per lane)
        for (int idx = lane; idx < 512; idx += 32) {
            const int r = idx >> 5, c = (idx & 31) * 4;
            const float4 vv = *(const float4*)(h + (size_t)(base + r) * 128 + c);
            _Float16* dst = my + r * HA_STRIDE + c;
            dst[0] = (_Float16)vv.x; dst[1] = (_Float16)vv.y;
            dst[2] = (_Float16)vv.z; dst[3] = (_Float16)vv.w;
        }
    }
    __syncthreads();

    if (active) {
        const int col = lane & 15, hh = lane >> 4;
        float psum[8];
#pragma unroll
        for (int v = 0; v < 8; ++v) psum[v] = 0.0f;

        for (int ntc = 0; ntc < 4; ++ntc) {       // N = 64
            v8f acc = {};
#pragma unroll
            for (int kt = 0; kt < 4; ++kt) {      // K = 128
                const v16h b = load_b_frag(pW3, 4, kt, ntc);
                const v16h a = load_a_frag(my, HA_STRIDE, 0, kt * 32);
                acc = wmma_f16(a, b, acc);
            }
            const int gc = ntc * 16 + col;
            const float bias = b3[gc], w4 = W4[gc];
#pragma unroll
            for (int v = 0; v < 8; ++v)
                psum[v] += fmaxf(acc[v] + bias, 0.0f) * w4;
        }
        // reduce over the 16 columns held by lanes within each half-wave
#pragma unroll
        for (int m = 1; m < 16; m <<= 1)
#pragma unroll
            for (int v = 0; v < 8; ++v)
                psum[v] += __shfl_xor(psum[v], m, 32);

        if (col == 0) {
            const float bb = b4[0];
#pragma unroll
            for (int v = 0; v < 8; ++v) {
                const float y = psum[v] + bb;
                out[base + v + 8 * hh] = 1.0f / (1.0f + __expf(-y));
            }
        }
    }
}

// ---------------------------------------------------------------------------
// Launch
// ---------------------------------------------------------------------------
extern "C" void kernel_launch(void* const* d_in, const int* in_sizes, int n_in,
                              void* d_out, int out_size, void* d_ws, size_t ws_size,
                              hipStream_t stream) {
    const float* x   = (const float*)d_in[0];
    const int*   ei  = (const int*)d_in[1];
    const float* W1a = (const float*)d_in[2];
    const float* b1a = (const float*)d_in[3];
    const float* W2a = (const float*)d_in[4];
    const float* b2a = (const float*)d_in[5];
    const float* W1b = (const float*)d_in[6];
    const float* b1b = (const float*)d_in[7];
    const float* W2b = (const float*)d_in[8];
    const float* b2b = (const float*)d_in[9];
    const float* W3  = (const float*)d_in[10];
    const float* b3  = (const float*)d_in[11];
    const float* W4  = (const float*)d_in[12];
    const float* b4  = (const float*)d_in[13];

    const int Nn = in_sizes[0] / 128;   // 10000
    const int E  = in_sizes[1] / 2;     // 320000

    char* ws = (char*)d_ws;
    size_t o = 0;
    _Float16* pW1a = (_Float16*)(ws + o); o += (size_t)256 * 512 * 2;
    _Float16* pW2a = (_Float16*)(ws + o); o += (size_t)512 * 256 * 2;
    _Float16* pW1b = (_Float16*)(ws + o); o += (size_t)512 * 256 * 2;
    _Float16* pW2b = (_Float16*)(ws + o); o += (size_t)256 * 128 * 2;
    _Float16* pW3  = (_Float16*)(ws + o); o += (size_t)128 * 64 * 2;
    o = (o + 255) & ~(size_t)255;
    float* agg1 = (float*)(ws + o); o += (size_t)Nn * 256 * 4;
    float* agg2 = (float*)(ws + o); o += (size_t)Nn * 128 * 4;

    // pack weights into WMMA B-fragment layout (f16)
    pack_weight_kernel<<<(256 * 512 + 255) / 256, 256, 0, stream>>>(W1a, pW1a, 256, 512);
    pack_weight_kernel<<<(512 * 256 + 255) / 256, 256, 0, stream>>>(W2a, pW2a, 512, 256);
    pack_weight_kernel<<<(512 * 256 + 255) / 256, 256, 0, stream>>>(W1b, pW1b, 512, 256);
    pack_weight_kernel<<<(256 * 128 + 255) / 256, 256, 0, stream>>>(W2b, pW2b, 256, 128);
    pack_weight_kernel<<<(128 * 64 + 255) / 256, 256, 0, stream>>>(W3, pW3, 128, 64);

    // zero aggregation buffers (ReLU outputs >= 0, so 0 == reference "empty node")
    zero_f32_kernel<<<(Nn * 256 + 255) / 256, 256, 0, stream>>>(agg1, Nn * 256);
    zero_f32_kernel<<<(Nn * 128 + 255) / 256, 256, 0, stream>>>(agg2, Nn * 128);

    edgeconv1_kernel<<<(E + 31) / 32, 256, 0, stream>>>(x, ei, E, pW1a, b1a, pW2a, b2a, agg1);
    edgeconv2_kernel<<<(E + 31) / 32, 256, 0, stream>>>(agg1, ei, E, pW1b, b1b, pW2b, b2b, agg2);
    head_kernel<<<(Nn + 127) / 128, 256, 0, stream>>>(agg2, pW3, b3, W4, b4, (float*)d_out, Nn);
}
